// Transformer_5463198401130
// MI455X (gfx1250) — matrix operand order repaired and checked by reading, not compiled, not run on hardware
//
#include <hip/hip_runtime.h>
#include <math.h>

typedef __attribute__((ext_vector_type(16))) __bf16 bf16x16;
typedef __attribute__((ext_vector_type(8)))  __bf16 bf16x8;
typedef __attribute__((ext_vector_type(8)))  float  f32x8;
typedef __attribute__((ext_vector_type(4)))  int    v4i;

#if __has_builtin(__builtin_amdgcn_global_load_async_to_lds_b128) && \
    __has_builtin(__builtin_amdgcn_s_wait_asynccnt)
#define HAVE_ASYNC_LDS 1
#endif

// ---------------- WMMA helpers (gfx1250 wave32, v_wmma_f32_16x16x32_bf16) ----------------
static __device__ __forceinline__ f32x8 wmma_bf(bf16x16 a, bf16x16 b, f32x8 c) {
  return __builtin_amdgcn_wmma_f32_16x16x32_bf16(false, a, false, b, (short)0, c, false, false);
}

// A fragment (16x32 bf16) from a row-major bf16 tile (LDS or global).
// ISA layout: lane l (m=l&15, kh=l>>4) needs A[m][kh*8 .. +7] and A[m][kh*8+16 .. +7]
// -> two contiguous 16-byte loads per lane.
static __device__ __forceinline__ bf16x16 ldA_row(const __bf16* A, int lda) {
  int l = threadIdx.x & 31, m = l & 15, kh = l >> 4;
  const __bf16* p = A + m * lda + kh * 8;
  bf16x8 lo = *(const bf16x8*)(p);
  bf16x8 hi = *(const bf16x8*)(p + 16);
  return __builtin_shufflevector(lo, hi, 0, 1, 2, 3, 4, 5, 6, 7, 8, 9, 10, 11, 12, 13, 14, 15);
}

// B fragment from pre-swizzled bf16 weights: fragment base + lane*16, one 32B load.
static __device__ __forceinline__ bf16x16 ldB_swz(const __bf16* Wfrag) {
  return *(const bf16x16*)(Wfrag + (size_t)(threadIdx.x & 31) * 16);
}

// ---------------- weight pre-swizzle: W (KxN f32 row-major) -> fragment-major bf16 ----------
// out idx = ((nt*nKt + kt)*32 + lane)*16 + v ; value = W[kt*32 + (lane>>4)*8 + (v&7) + (v>>3)*16][nt*16 + (lane&15)]
__global__ __launch_bounds__(256) void k_prep_w(const float* __restrict__ W,
                                                __bf16* __restrict__ Wb, int K, int N) {
  int idx = blockIdx.x * 256 + threadIdx.x;
  if (idx >= K * N) return;
  int v = idx & 15;
  int lane = (idx >> 4) & 31;
  int frag = idx >> 9;
  int nKt = K >> 5;
  int kt = frag % nKt;
  int nt = frag / nKt;
  int n = lane & 15, kh = lane >> 4;
  int k = kt * 32 + kh * 8 + (v & 7) + (v >> 3) * 16;
  Wb[idx] = (__bf16)W[(size_t)k * N + nt * 16 + n];
}

__global__ __launch_bounds__(256) void k_cvt_bf16(const float* __restrict__ A,
                                                  __bf16* __restrict__ B, int nElem) {
  int idx = blockIdx.x * 256 + threadIdx.x;
  if (idx < nElem) B[idx] = (__bf16)A[idx];
}

// ---------------- LayerNorm helpers (256-thread blocks) ----------------
static __device__ __forceinline__ void ln64(
    const float* sA, const float* sB, const float* g, const float* bl,
    float* sRs, float* sRq, float* sOutF, __bf16* sOutB, float* gOut, int gStride)
{
  int t = threadIdx.x, r = t >> 4, q = t & 15;
  float vals[4]; float s = 0.f, sq = 0.f;
#pragma unroll
  for (int u = 0; u < 4; ++u) {
    int cc = q * 4 + u;
    float v = sA[r * 64 + cc] + sB[r * 64 + cc];
    vals[u] = v; s += v; sq += v * v;
  }
  sRs[r * 16 + q] = s; sRq[r * 16 + q] = sq;
  __syncthreads();
  float S = 0.f, SQ = 0.f;
#pragma unroll
  for (int k = 0; k < 16; ++k) { S += sRs[r * 16 + k]; SQ += sRq[r * 16 + k]; }
  float mean = S * (1.f / 64.f);
  float inv  = rsqrtf(fmaxf(SQ * (1.f / 64.f) - mean * mean, 0.f) + 1e-5f);
#pragma unroll
  for (int u = 0; u < 4; ++u) {
    int cc = q * 4 + u;
    float v = (vals[u] - mean) * inv * g[cc] + bl[cc];
    if (sOutF) sOutF[r * 64 + cc] = v;
    if (sOutB) sOutB[r * 64 + cc] = (__bf16)v;
    if (gOut)  gOut[(size_t)r * gStride + cc] = v;
  }
  __syncthreads();
}

static __device__ __forceinline__ void ln256(
    const float* sIn, const float* g, const float* bl,
    float* sRs, float* sRq, float* sOutF, __bf16* sOutB, float* gOut)
{
  int t = threadIdx.x, r = t >> 4, q = t & 15;
  float vals[16]; float s = 0.f, sq = 0.f;
#pragma unroll
  for (int u = 0; u < 16; ++u) {
    float v = sIn[r * 256 + q * 16 + u];
    vals[u] = v; s += v; sq += v * v;
  }
  sRs[r * 16 + q] = s; sRq[r * 16 + q] = sq;
  __syncthreads();
  float S = 0.f, SQ = 0.f;
#pragma unroll
  for (int k = 0; k < 16; ++k) { S += sRs[r * 16 + k]; SQ += sRq[r * 16 + k]; }
  float mean = S * (1.f / 256.f);
  float inv  = rsqrtf(fmaxf(SQ * (1.f / 256.f) - mean * mean, 0.f) + 1e-5f);
#pragma unroll
  for (int u = 0; u < 16; ++u) {
    int cc = q * 16 + u;
    float v = (vals[u] - mean) * inv * g[cc] + bl[cc];
    if (sOutF) sOutF[r * 256 + cc] = v;
    if (sOutB) sOutB[r * 256 + cc] = (__bf16)v;
    if (gOut)  gOut[(size_t)r * 256 + cc] = v;
  }
  __syncthreads();
}

// ---------------- K0a: zX stats over nodes (mean/min/max/std ddof=1) ----------------
__global__ __launch_bounds__(256) void k0a_zx(const float* __restrict__ X, float* __restrict__ zX) {
  const int b = blockIdx.x, c = threadIdx.x;
  float s = 0.f, sq = 0.f, mn = 1e30f, mx = -1e30f;
  for (int i = 0; i < 256; ++i) {
    float v = X[((size_t)b * 256 + i) * 256 + c];
    s += v; sq += v * v; mn = fminf(mn, v); mx = fmaxf(mx, v);
  }
  zX[b * 1024 + c]       = s * (1.f / 256.f);
  zX[b * 1024 + 256 + c] = mn;
  zX[b * 1024 + 512 + c] = mx;
  zX[b * 1024 + 768 + c] = sqrtf(fmaxf((sq - s * s * (1.f / 256.f)) * (1.f / 255.f), 0.f));
}

// ---------------- K0b: zE stats over (i,j) ----------------
__global__ __launch_bounds__(256) void k0b_ze(const float* __restrict__ E, float* __restrict__ zE) {
  const int c = blockIdx.x, b = blockIdx.y, t = threadIdx.x;
  __shared__ float rs[256], rq[256], rmn[256], rmx[256];
  float s = 0.f, sq = 0.f, mn = 1e30f, mx = -1e30f;
  for (int u = t; u < 65536; u += 256) {
    float v = E[((size_t)b * 65536 + u) * 64 + c];
    s += v; sq += v * v; mn = fminf(mn, v); mx = fmaxf(mx, v);
  }
  rs[t] = s; rq[t] = sq; rmn[t] = mn; rmx[t] = mx;
  __syncthreads();
  for (int off = 128; off > 0; off >>= 1) {
    if (t < off) {
      rs[t] += rs[t + off]; rq[t] += rq[t + off];
      rmn[t] = fminf(rmn[t], rmn[t + off]); rmx[t] = fmaxf(rmx[t], rmx[t + off]);
    }
    __syncthreads();
  }
  if (t == 0) {
    float S = rs[0];
    zE[b * 256 + c]       = S * (1.f / 65536.f);
    zE[b * 256 + 64 + c]  = rmn[0];
    zE[b * 256 + 128 + c] = rmx[0];
    zE[b * 256 + 192 + c] = sqrtf(fmaxf((rq[0] - S * S * (1.f / 65536.f)) * (1.f / 65535.f), 0.f));
  }
}

// ---------------- K1: Q/K/V projections, one wave per 16x16 output tile ----------------
__global__ __launch_bounds__(32) void k1_qkv(
    const __bf16* __restrict__ Xb, const unsigned char* __restrict__ nmk,
    const __bf16* __restrict__ Wqb, const float* __restrict__ bq,
    const __bf16* __restrict__ Wkb, const float* __restrict__ bk,
    const __bf16* __restrict__ Wvb, const float* __restrict__ bv,
    float* __restrict__ Qo, float* __restrict__ Ko, float* __restrict__ Vo)
{
  const int mt = blockIdx.x, nt = blockIdx.y, sel = blockIdx.z;
  const __bf16* W = (sel == 0) ? Wqb : (sel == 1) ? Wkb : Wvb;
  const float* B  = (sel == 0) ? bq : (sel == 1) ? bk : bv;
  float* O        = (sel == 0) ? Qo : (sel == 1) ? Ko : Vo;
  f32x8 c = {};
#pragma unroll
  for (int ks = 0; ks < 8; ++ks) {
    bf16x16 a = ldA_row(Xb + (size_t)mt * 16 * 256 + ks * 32, 256);
    bf16x16 b = ldB_swz(W + ((size_t)nt * 8 + ks) * 512);
    c = wmma_bf(a, b, c);
  }
  int l = threadIdx.x & 31, nn = l & 15, mh = l >> 4;
#pragma unroll
  for (int r = 0; r < 8; ++r) {
    int m = r + 8 * mh, g = mt * 16 + m;
    int bb = g >> 8, ii = g & 255;
    float xm = nmk[bb * 256 + ii] ? 1.f : 0.f;
    int col = nt * 16 + nn;
    O[(size_t)g * 256 + col] = (c[r] + B[col]) * xm;
  }
}

// ---------------- K0c: y path + FiLM vectors (small, scalar) ----------------
__global__ __launch_bounds__(256) void k0c_ypath(
    const float* __restrict__ y, const float* __restrict__ zX, const float* __restrict__ zE,
    const float* __restrict__ Wyem, const float* __restrict__ byem,
    const float* __restrict__ Wyea, const float* __restrict__ byea,
    const float* __restrict__ Wyxm, const float* __restrict__ byxm,
    const float* __restrict__ Wyxa, const float* __restrict__ byxa,
    const float* __restrict__ Wyy, const float* __restrict__ byy,
    const float* __restrict__ Wxy, const float* __restrict__ bxy,
    const float* __restrict__ Wey, const float* __restrict__ bey,
    const float* __restrict__ Wo1, const float* __restrict__ bo1,
    const float* __restrict__ Wo2, const float* __restrict__ bo2,
    const float* __restrict__ Wf1, const float* __restrict__ bf1,
    const float* __restrict__ Wf2, const float* __restrict__ bf2,
    const float* __restrict__ g1, const float* __restrict__ bl1,
    const float* __restrict__ g2, const float* __restrict__ bl2,
    float* __restrict__ ye1, float* __restrict__ ye2,
    float* __restrict__ yx1, float* __restrict__ yx2,
    float* __restrict__ yout)
{
  const int b = blockIdx.x, t = threadIdx.x;
  __shared__ float sy[64], sv[64], sv2[64], sn[64], syr[64], sh[2048];
  if (t < 64) sy[t] = y[b * 64 + t];
  __syncthreads();
  { // FiLM projections (y_e_add, y_e_mul, y_x_add, y_x_mul)
    float a1 = byea[t], a2 = byem[t], a3 = byxa[t], a4 = byxm[t];
    for (int k = 0; k < 64; ++k) {
      float yv = sy[k];
      a1 += yv * Wyea[k * 256 + t];
      a2 += yv * Wyem[k * 256 + t];
      a3 += yv * Wyxa[k * 256 + t];
      a4 += yv * Wyxm[k * 256 + t];
    }
    ye1[b * 256 + t] = a1; ye2[b * 256 + t] = a2;
    yx1[b * 256 + t] = a3; yx2[b * 256 + t] = a4;
  }
  if (t < 64) {
    float a = byy[t] + bxy[t] + bey[t];
    for (int k = 0; k < 64; ++k)   a += sy[k] * Wyy[k * 64 + t];
    for (int k = 0; k < 1024; ++k) a += zX[b * 1024 + k] * Wxy[k * 64 + t];
    for (int k = 0; k < 256; ++k)  a += zE[b * 256 + k] * Wey[k * 64 + t];
    sv[t] = a;
  }
  __syncthreads();
  if (t < 64) {
    float a = bo1[t];
    for (int k = 0; k < 64; ++k) a += sv[k] * Wo1[k * 64 + t];
    sv2[t] = fmaxf(a, 0.f);
  }
  __syncthreads();
  if (t < 64) {
    float a = bo2[t];
    for (int k = 0; k < 64; ++k) a += sv2[k] * Wo2[k * 64 + t];
    sn[t] = sy[t] + a;  // y + new_y
  }
  __syncthreads();
  if (t < 64) { // yr = LN64(y + new_y)
    float S = 0.f, SQ = 0.f;
    for (int k = 0; k < 64; ++k) { float v = sn[k]; S += v; SQ += v * v; }
    float mean = S * (1.f / 64.f);
    float inv  = rsqrtf(fmaxf(SQ * (1.f / 64.f) - mean * mean, 0.f) + 1e-5f);
    syr[t] = (sn[t] - mean) * inv * g1[t] + bl1[t];
  }
  __syncthreads();
  for (int u = 0; u < 8; ++u) { // ffy1 + relu
    int col = u * 256 + t;
    float a = bf1[col];
    for (int k = 0; k < 64; ++k) a += syr[k] * Wf1[k * 2048 + col];
    sh[col] = fmaxf(a, 0.f);
  }
  __syncthreads();
  if (t < 64) { // ffy2 + residual
    float a = bf2[t];
    for (int k = 0; k < 2048; ++k) a += sh[k] * Wf2[k * 64 + t];
    sv[t] = syr[t] + a;
  }
  __syncthreads();
  if (t < 64) { // final LN64
    float S = 0.f, SQ = 0.f;
    for (int k = 0; k < 64; ++k) { float v = sv[k]; S += v; SQ += v * v; }
    float mean = S * (1.f / 64.f);
    float inv  = rsqrtf(fmaxf(SQ * (1.f / 64.f) - mean * mean, 0.f) + 1e-5f);
    yout[b * 64 + t] = (sv[t] - mean) * inv * g2[t] + bl2[t];
  }
}

// ---------------- K2: fused E/attention pipeline, one workgroup per (b,i) row ----------------
__global__ __launch_bounds__(256) void k2_edge(
    const float* __restrict__ E, const unsigned char* __restrict__ nmk,
    const float* __restrict__ Qp, const float* __restrict__ Kp, const float* __restrict__ Vp,
    const __bf16* __restrict__ Wem, const float* __restrict__ bem,   // e_mul (swizzled)
    const __bf16* __restrict__ Wea, const float* __restrict__ bea,   // e_add (swizzled)
    const __bf16* __restrict__ Weo, const float* __restrict__ beo,   // e_out (swizzled)
    const __bf16* __restrict__ Wf1, const float* __restrict__ bf1,   // ffE1 (swizzled)
    const __bf16* __restrict__ Wf2, const float* __restrict__ bf2,   // ffE2 (swizzled)
    const float* __restrict__ gE1, const float* __restrict__ blE1,
    const float* __restrict__ gE2, const float* __restrict__ blE2,
    const float* __restrict__ ye1, const float* __restrict__ ye2,
    const float* __restrict__ yx1, const float* __restrict__ yx2,
    float* __restrict__ Eout, float* __restrict__ Xpre)
{
  __shared__ float  sEf[16 * 64];
  __shared__ __align__(32) __bf16 sEb[16 * 64];
  __shared__ float  sY[16 * 256];
  __shared__ __align__(32) __bf16 sNE[16 * 256];
  __shared__ float  sNO[16 * 64];
  __shared__ float  sErf[16 * 64];
  __shared__ __align__(32) __bf16 sErb[16 * 64];
  __shared__ __align__(32) __bf16 sH[16 * 128];
  __shared__ float  sF[16 * 64];
  __shared__ float  sQ[256];
  __shared__ float  sYE1[256];
  __shared__ float  sYE2[256];
  __shared__ float  sRs[256];
  __shared__ float  sRq[256];
  // hoisted loop-invariants (masks / biases / LN params)
  __shared__ float  sMk[256];
  __shared__ float  sBem[256], sBea[256];
  __shared__ float  sBeo[64], sBf2[64], sBf1[128];
  __shared__ float  sLg1[64], sLb1[64], sLg2[64], sLb2[64];

  const int i = blockIdx.x, b = blockIdx.y;
  const int t = threadIdx.x, w = t >> 5;
  const int l = t & 31, nn = l & 15, mh = l >> 4;
  const float xi = nmk[b * 256 + i] ? 1.f : 0.f;
  const size_t rowI = ((size_t)b * 256 + i) * 256;
  const size_t rowE = ((size_t)b * 256 + i) * 256;

  sQ[t]   = Qp[rowI + t];
  sYE1[t] = ye1[b * 256 + t];
  sYE2[t] = ye2[b * 256 + t];
  sMk[t]  = nmk[b * 256 + t] ? 1.f : 0.f;
  sBem[t] = bem[t];
  sBea[t] = bea[t];
  if (t < 64) {
    sBeo[t] = beo[t]; sBf2[t] = bf2[t];
    sLg1[t] = gE1[t]; sLb1[t] = blE1[t];
    sLg2[t] = gE2[t]; sLb2[t] = blE2[t];
  }
  if (t < 128) sBf1[t] = bf1[t];

  float mRun = -1e30f, sRun = 0.f, aRun = 0.f;  // online softmax state, column c = t

  for (int jt = 0; jt < 16; ++jt) {
    __syncthreads();
    // ---- stage E tile (16 rows of 64) ----
#ifdef HAVE_ASYNC_LDS
    // async global->LDS DMA (bypasses VGPRs, tracked by ASYNCcnt)
    __builtin_amdgcn_global_load_async_to_lds_b128(
        (v4i*)(E + (rowE + jt * 16) * 64 + t * 4),
        (v4i*)(sEf + t * 4), 0, 0);
    if (jt < 15)
      __builtin_prefetch(&E[(rowE + (jt + 1) * 16) * 64 + t * 4], 0, 1);
    __builtin_amdgcn_s_wait_asynccnt(0);
    __syncthreads();
#pragma unroll
    for (int u = 0; u < 4; ++u) sEb[t * 4 + u] = (__bf16)sEf[t * 4 + u];
#else
    {
      int u0 = t * 4;
#pragma unroll
      for (int u = 0; u < 4; ++u) {
        int idx = u0 + u;
        float v = E[(rowE + jt * 16) * 64 + idx];
        sEf[idx] = v; sEb[idx] = (__bf16)v;
      }
      if (jt < 15)
        __builtin_prefetch(&E[(rowE + (jt + 1) * 16) * 64 + t * 4], 0, 1);
    }
#endif
    __syncthreads();

    // E1/E2 GEMMs (K=64), fuse qk modulation -> Y, FiLM -> newE (bf16)
#pragma unroll
    for (int p = 0; p < 2; ++p) {
      int nt = 2 * w + p;
      f32x8 c1 = {}; f32x8 c2 = {};
#pragma unroll
      for (int ks = 0; ks < 2; ++ks) {
        bf16x16 a = ldA_row(sEb + ks * 32, 64);
        c1 = wmma_bf(a, ldB_swz(Wem + ((size_t)nt * 2 + ks) * 512), c1);
        c2 = wmma_bf(a, ldB_swz(Wea + ((size_t)nt * 2 + ks) * 512), c2);
      }
#pragma unroll
      for (int r = 0; r < 8; ++r) {
        int mm = r + 8 * mh, j = jt * 16 + mm, cN = nt * 16 + nn;
        float em = xi * sMk[j];
        float e1 = (c1[r] + sBem[cN]) * em;
        float e2 = (c2[r] + sBea[cN]) * em;
        float qk = sQ[cN] * Kp[((size_t)b * 256 + j) * 256 + cN] * 0.17677669529663687f;
        float yv = qk * (e1 + 1.f) + e2;
        sY[mm * 256 + cN]  = yv;
        sNE[mm * 256 + cN] = (__bf16)(sYE1[cN] + (sYE2[cN] + 1.f) * yv);
      }
    }
    __syncthreads();

    // online softmax over this j-tile (thread owns column t)
#pragma unroll
    for (int r = 0; r < 16; ++r) {
      int j = jt * 16 + r;
      if (sMk[j] != 0.f) {
        float yv = sY[r * 256 + t];
        float mN = fmaxf(mRun, yv);
        float sc = __expf(mRun - mN);
        float pv = __expf(yv - mN);
        sRun = sRun * sc + pv;
        aRun = aRun * sc + pv * Vp[((size_t)b * 256 + j) * 256 + t];
        mRun = mN;
      }
    }

    // e_out: newE(16x256) @ Weo(256x64)  (waves 0..3)
    if (w < 4) {
      f32x8 c = {};
#pragma unroll
      for (int ks = 0; ks < 8; ++ks)
        c = wmma_bf(ldA_row(sNE + ks * 32, 256), ldB_swz(Weo + ((size_t)w * 8 + ks) * 512), c);
#pragma unroll
      for (int r = 0; r < 8; ++r) {
        int mm = r + 8 * mh, j = jt * 16 + mm, cc = w * 16 + nn;
        sNO[mm * 64 + cc] = (c[r] + sBeo[cc]) * (xi * sMk[j]);
      }
    }
    __syncthreads();

    // Er = LN64(E + newE_out) -> f32 + bf16
    ln64(sEf, sNO, sLg1, sLb1, sRs, sRq, sErf, sErb, (float*)0, 0);

    // ffE1: Er(16x64)@Wf1(64x128) + relu  (8 waves, n-tile = w)
    {
      f32x8 c = {};
#pragma unroll
      for (int ks = 0; ks < 2; ++ks)
        c = wmma_bf(ldA_row(sErb + ks * 32, 64), ldB_swz(Wf1 + ((size_t)w * 2 + ks) * 512), c);
#pragma unroll
      for (int r = 0; r < 8; ++r) {
        int mm = r + 8 * mh, cc = w * 16 + nn;
        sH[mm * 128 + cc] = (__bf16)fmaxf(c[r] + sBf1[cc], 0.f);
      }
    }
    __syncthreads();

    // ffE2: h(16x128)@Wf2(128x64)  (waves 0..3)
    if (w < 4) {
      f32x8 c = {};
#pragma unroll
      for (int ks = 0; ks < 4; ++ks)
        c = wmma_bf(ldA_row(sH + ks * 32, 128), ldB_swz(Wf2 + ((size_t)w * 4 + ks) * 512), c);
#pragma unroll
      for (int r = 0; r < 8; ++r) {
        int mm = r + 8 * mh, cc = w * 16 + nn;
        sF[mm * 64 + cc] = c[r] + sBf2[cc];
      }
    }
    __syncthreads();

    // E_out = LN64(Er + ff) -> global (written exactly once)
    ln64(sErf, sF, sLg2, sLb2, sRs, sRq, (float*)0, (__bf16*)0,
         Eout + (rowE + jt * 16) * 64, 64);
  }

  // finalize attention + FiLM -> Xpre
  {
    float wv = (sRun > 0.f) ? (aRun / sRun) : 0.f;
    Xpre[rowI + t] = yx1[b * 256 + t] + (yx2[b * 256 + t] + 1.f) * wv;
  }
}

// ---------------- K3: X path (x_out -> LN -> ffX1/ffX2 K-chunked -> LN) ----------------
__global__ __launch_bounds__(256) void k3_xpath(
    const float* __restrict__ X, const unsigned char* __restrict__ nmk,
    const float* __restrict__ Xpre,
    const __bf16* __restrict__ Wxo, const float* __restrict__ bxo,
    const __bf16* __restrict__ W1, const float* __restrict__ b1,
    const __bf16* __restrict__ W2, const float* __restrict__ b2,
    const float* __restrict__ gX1, const float* __restrict__ blX1,
    const float* __restrict__ gX2, const float* __restrict__ blX2,
    float* __restrict__ Xout)
{
  __shared__ __align__(32) __bf16 sAb[16 * 256];
  __shared__ float  sT[16 * 256];
  __shared__ float  sXrF[16 * 256];
  __shared__ __align__(32) __bf16 sXrB[16 * 256];
  __shared__ __align__(32) __bf16 sH[16 * 128];
  __shared__ float  sRs[256], sRq[256];

  const int mt = blockIdx.x;
  const int t = threadIdx.x, w = t >> 5;
  const int l = t & 31, nn = l & 15, mh = l >> 4;

#pragma unroll
  for (int u = 0; u < 16; ++u) {
    int idx = t + u * 256;
    sAb[idx] = (__bf16)Xpre[(size_t)mt * 16 * 256 + idx];
  }
  __syncthreads();

  // x_out (K=256) + mask + residual
#pragma unroll
  for (int p = 0; p < 2; ++p) {
    int nt = 2 * w + p;
    f32x8 c = {};
#pragma unroll
    for (int ks = 0; ks < 8; ++ks)
      c = wmma_bf(ldA_row(sAb + ks * 32, 256), ldB_swz(Wxo + ((size_t)nt * 8 + ks) * 512), c);
#pragma unroll
    for (int r = 0; r < 8; ++r) {
      int mm = r + 8 * mh, g = mt * 16 + mm;
      int bb = g >> 8, ii = g & 255;
      float xm = nmk[bb * 256 + ii] ? 1.f : 0.f;
      int col = nt * 16 + nn;
      sT[mm * 256 + col] = (c[r] + bxo[col]) * xm + X[(size_t)g * 256 + col];
    }
  }
  __syncthreads();

  ln256(sT, gX1, blX1, sRs, sRq, sXrF, sXrB, (float*)0);

  // ffX1(256->2048, relu) / ffX2(2048->256) with K-chunked WMMA accumulation in registers
  f32x8 acc0 = {}; f32x8 acc1 = {};
  for (int kc = 0; kc < 16; ++kc) {
    {
      int hnt = kc * 8 + w;            // ffX1 output n-tile (of 128)
      int hcol = hnt * 16;
      f32x8 c = {};
#pragma unroll
      for (int ks = 0; ks < 8; ++ks)
        c = wmma_bf(ldA_row(sXrB + ks * 32, 256), ldB_swz(W1 + ((size_t)hnt * 8 + ks) * 512), c);
#pragma unroll
      for (int r = 0; r < 8; ++r) {
        int mm = r + 8 * mh;
        sH[mm * 128 + w * 16 + nn] = (__bf16)fmaxf(c[r] + b1[hcol + nn], 0.f);
      }
    }
    __syncthreads();
#pragma unroll
    for (int ks = 0; ks < 4; ++ks) {
      bf16x16 a = ldA_row(sH + ks * 32, 128);
      int kt = kc * 4 + ks;            // ffX2 K-tile (of 64)
      acc0 = wmma_bf(a, ldB_swz(W2 + ((size_t)(2 * w) * 64 + kt) * 512), acc0);
      acc1 = wmma_bf(a, ldB_swz(W2 + ((size_t)(2 * w + 1) * 64 + kt) * 512), acc1);
    }
    __syncthreads();
  }

#pragma unroll
  for (int r = 0; r < 8; ++r) {
    int mm = r + 8 * mh;
    int c0 = (2 * w) * 16 + nn, c1 = (2 * w + 1) * 16 + nn;
    sT[mm * 256 + c0] = acc0[r] + b2[c0] + sXrF[mm * 256 + c0];
    sT[mm * 256 + c1] = acc1[r] + b2[c1] + sXrF[mm * 256 + c1];
  }
  __syncthreads();

  ln256(sT, gX2, blX2, sRs, sRq, (float*)0, (__bf16*)0, Xout + (size_t)mt * 16 * 256);
}

// ---------------- host launcher ----------------
extern "C" void kernel_launch(void* const* d_in, const int* in_sizes, int n_in,
                              void* d_out, int out_size, void* d_ws, size_t ws_size,
                              hipStream_t stream) {
  (void)in_sizes; (void)n_in; (void)out_size; (void)ws_size;
  const float* X = (const float*)d_in[0];
  const float* E = (const float*)d_in[1];
  const float* y = (const float*)d_in[2];
  const unsigned char* nmk = (const unsigned char*)d_in[3];
#define W(i) ((const float*)d_in[i])
  float* ws   = (float*)d_ws;
  float* Qb   = ws;                 // 262144 each
  float* Kb   = Qb + 262144;
  float* Vb   = Kb + 262144;
  float* Xpre = Vb + 262144;
  float* ye1  = Xpre + 262144;      // 1024 each
  float* ye2  = ye1 + 1024;
  float* yx1  = ye2 + 1024;
  float* yx2  = yx1 + 1024;
  float* zX   = yx2 + 1024;         // 4096
  float* zE   = zX + 4096;          // 1024 -> f32 total 1,057,792 (32B aligned)

  __bf16* wb  = (__bf16*)(ws + 1057792);
  __bf16* QWb = wb;                  // 65536
  __bf16* KWb = wb + 65536;          // 65536
  __bf16* VWb = wb + 131072;         // 65536
  __bf16* EMb = wb + 196608;         // 16384
  __bf16* EAb = wb + 212992;         // 16384
  __bf16* EOb = wb + 229376;         // 16384
  __bf16* F1E = wb + 245760;         // 8192
  __bf16* F2E = wb + 253952;         // 8192
  __bf16* XOb = wb + 262144;         // 65536
  __bf16* F1X = wb + 327680;         // 524288
  __bf16* F2X = wb + 851968;         // 524288
  __bf16* Xbf = wb + 1376256;        // 262144

  float* Xout = (float*)d_out;
  float* Eout = Xout + 262144;
  float* yout = Eout + 16777216;

  // weight pre-swizzle (f32 -> bf16 fragment-major; tiny vs 150MB main traffic)
  k_prep_w<<<256, 256, 0, stream>>>(W(4),  QWb, 256, 256);
  k_prep_w<<<256, 256, 0, stream>>>(W(6),  KWb, 256, 256);
  k_prep_w<<<256, 256, 0, stream>>>(W(8),  VWb, 256, 256);
  k_prep_w<<<64,  256, 0, stream>>>(W(12), EMb, 64, 256);
  k_prep_w<<<64,  256, 0, stream>>>(W(10), EAb, 64, 256);
  k_prep_w<<<64,  256, 0, stream>>>(W(30), EOb, 256, 64);
  k_prep_w<<<32,  256, 0, stream>>>(W(40), F1E, 64, 128);
  k_prep_w<<<32,  256, 0, stream>>>(W(42), F2E, 128, 64);
  k_prep_w<<<256, 256, 0, stream>>>(W(28), XOb, 256, 256);
  k_prep_w<<<2048,256, 0, stream>>>(W(36), F1X, 256, 2048);
  k_prep_w<<<2048,256, 0, stream>>>(W(38), F2X, 2048, 256);
  k_cvt_bf16<<<1024, 256, 0, stream>>>(X, Xbf, 262144);

  k0a_zx<<<4, 256, 0, stream>>>(X, zX);
  k0b_ze<<<dim3(64, 4), 256, 0, stream>>>(E, zE);
  k1_qkv<<<dim3(64, 16, 3), 32, 0, stream>>>(Xbf, nmk, QWb, W(5), KWb, W(7), VWb, W(9),
                                             Qb, Kb, Vb);
  k0c_ypath<<<4, 256, 0, stream>>>(y, zX, zE,
      W(14), W(15), W(16), W(17), W(18), W(19), W(20), W(21),
      W(22), W(23), W(24), W(25), W(26), W(27),
      W(32), W(33), W(34), W(35), W(44), W(45), W(46), W(47),
      W(56), W(57), W(58), W(59),
      ye1, ye2, yx1, yx2, yout);
  k2_edge<<<dim3(256, 4), 256, 0, stream>>>(E, nmk, Qb, Kb, Vb,
      EMb, W(13), EAb, W(11), EOb, W(31),
      F1E, W(41), F2E, W(43), W(52), W(53), W(54), W(55),
      ye1, ye2, yx1, yx2, Eout, Xpre);
  k3_xpath<<<64, 256, 0, stream>>>(X, nmk, Xpre,
      XOb, W(29), F1X, W(37), F2X, W(39),
      W(48), W(49), W(50), W(51), Xout);
#undef W
}
